// QuantShiftConvNet_81501299409069
// MI455X (gfx1250) — matrix operand order repaired and checked by reading, not compiled, not run on hardware
//
#include <hip/hip_runtime.h>
#include <cstdint>
#include <cstddef>

typedef __attribute__((ext_vector_type(16))) _Float16 v16h;
typedef __attribute__((ext_vector_type(8)))  float    v8f;
typedef __attribute__((ext_vector_type(4)))  unsigned int v4u;
typedef __attribute__((ext_vector_type(8)))  int      v8i;
typedef __attribute__((ext_vector_type(4)))  int      v4i;

// ---------------------------------------------------------------------------
// Helpers
// ---------------------------------------------------------------------------
__device__ __forceinline__ float aq15(float x) {
  // DoReFa activation quantization: clip to [0,1], 15 levels, RNE
  x = fminf(fmaxf(x, 0.f), 1.f);
  return __builtin_rintf(x * 15.f) * (1.f / 15.f);
}

__device__ __forceinline__ float wq_val(float w, float mx) {
  // DoReFa weight quantization given per-tensor max|tanh(w)|
  float t = tanhf(w) / (2.f * mx) + 0.5f;
  t = __builtin_rintf(t * 15.f) * (1.f / 15.f);
  return 2.f * t - 1.f;
}

// A-fragment (16x32 f16) index inside a 512-half fragment.
// ISA layout: lane = row + 16*((k>>3)&1); slot = (k&7) | ((k>>4)<<3)
__device__ __forceinline__ int frag_a_idx(int r, int k) {
  int hf   = (k >> 3) & 1;
  int slot = (k & 7) | ((k >> 4) << 3);
  return (((r | (hf << 4))) << 4) | slot;
}

// B-fragment (32x16 f16): lane = col + 16*((k>>3)&1); slot = (k&7) | ((k>>4)<<3)
__device__ __forceinline__ int frag_b_idx(int n, int k) {
  int lane = n | (((k >> 3) & 1) << 4);
  return (lane << 4) | ((k & 7) | ((k >> 4) << 3));
}

// ---------------------------------------------------------------------------
// TDM: DMA one 1KB fragment (128 x 8B) global -> LDS.
// D# per CDNA5 ISA: group0 = {count=1, lds_addr,
// global_addr[56:0], type=2}; group1 = {mask=0, data_size=3(8B),
// tensor_dim0=128, tensor_dim1=1, tile_dim0=128, tile_dim1=1, stride0=128}.
// Issued once per wave (EXEC ignored); completion via TENSORcnt.
// 6-arg builtin on this toolchain:
//   (uint32x4 g0, int32x8 g1, int32x4 g2, int32x4 g3, int32x8, i32 cpol)
// ---------------------------------------------------------------------------
#if __has_builtin(__builtin_amdgcn_tensor_load_to_lds)
#define HAVE_TDM 1
#else
#define HAVE_TDM 0
#endif

__device__ __forceinline__ void stage_1k(const void* gsrc, _Float16* ldst,
                                         unsigned lds_byte_off) {
#if HAVE_TDM
  unsigned long long ga = (unsigned long long)(uintptr_t)gsrc;
  v4u g0;
  g0.x = 1u;                                        // count=1 valid descriptor
  g0.y = lds_byte_off;                              // lds_addr
  g0.z = (unsigned)ga;                              // global_addr[31:0]
  g0.w = (unsigned)((ga >> 32) & 0x1FFFFFFu) | (2u << 30);  // addr[56:32] | type=2
  v8i g1;
  g1[0] = 0x30000;            // workgroup_mask=0, data_size=3 (8B), flags=0
  g1[1] = (int)(128u << 16);  // tensor_dim0 = 128 (8B units)
  g1[2] = (int)(1u << 16);    // tensor_dim1 = 1
  g1[3] = (int)(128u << 16);  // tile_dim0 = 128
  g1[4] = 1;                  // tile_dim1 = 1, tile_dim2 = 0
  g1[5] = 128;                // tensor_dim0_stride = 128
  g1[6] = 0;
  g1[7] = 0;
  v4i z4 = {0, 0, 0, 0};
  v8i z8 = {0, 0, 0, 0, 0, 0, 0, 0};
  __builtin_amdgcn_tensor_load_to_lds(g0, g1, z4, z4, z8, 0);
  // Escape the LDS destination pointer into an asm with a memory clobber:
  // the TDM builtin only consumes the integer offset, so without this the
  // compiler proves the __shared__ array is never stored and folds the
  // subsequent ds loads to undef (observed: wmma fed v[0:7],v[0:7]).
  asm volatile("" :: "v"(ldst) : "memory");
#else
  // Fallback: the issuing wave copies cooperatively (32 lanes x 32B).
  int lane = threadIdx.x & 31;
  const uint4* s = (const uint4*)gsrc;
  uint4* d = (uint4*)ldst;
  d[lane * 2 + 0] = s[lane * 2 + 0];
  d[lane * 2 + 1] = s[lane * 2 + 1];
#endif
}

// Wait for DMA completion; memory clobber orders the LDS reads after it.
__device__ __forceinline__ void wait_tdm() {
#if HAVE_TDM
  __builtin_amdgcn_s_wait_tensorcnt(0);
#endif
  asm volatile("" ::: "memory");
}

// ---------------------------------------------------------------------------
// Small-weight prep: per-tensor max|tanh| reductions + pack B fragments
// ---------------------------------------------------------------------------
__global__ void k_prep_small(const float* __restrict__ w1, const float* __restrict__ w2,
                             const float* __restrict__ w3, const float* __restrict__ w4,
                             const float* __restrict__ fc2w,
                             float* __restrict__ wq1_out,
                             _Float16* __restrict__ B2pk, _Float16* __restrict__ B3pk,
                             _Float16* __restrict__ B4pk, _Float16* __restrict__ Bfc2pk) {
  __shared__ float red[256];
  __shared__ float mx[5];
  const float* tens[5] = {w1, w2, w3, w4, fc2w};
  const int    ns[5]   = {16, 512, 1024, 1024, 1280};
  int t = threadIdx.x;
  for (int i = 0; i < 5; ++i) {
    float m = 0.f;
    for (int j = t; j < ns[i]; j += 256) m = fmaxf(m, fabsf(tanhf(tens[i][j])));
    red[t] = m; __syncthreads();
    for (int s = 128; s > 0; s >>= 1) { if (t < s) red[t] = fmaxf(red[t], red[t + s]); __syncthreads(); }
    if (t == 0) mx[i] = red[0];
    __syncthreads();
  }
  if (t < 16) wq1_out[t] = wq_val(w1[t], mx[0]);
  for (int i = t; i < 1024; i += 256) {            // conv2: K padded 16->32
    int n = i >> 5, k = i & 31;
    float v = (k < 16) ? wq_val(w2[n * 16 + k], mx[1]) : 0.f;
    B2pk[((n >> 4) << 9) + frag_b_idx(n & 15, k)] = (_Float16)v;
  }
  for (int i = t; i < 1024; i += 256) {            // conv3
    int n = i >> 5, k = i & 31;
    B3pk[((n >> 4) << 9) + frag_b_idx(n & 15, k)] = (_Float16)wq_val(w3[n * 32 + k], mx[2]);
  }
  for (int i = t; i < 1024; i += 256) {            // conv4
    int n = i >> 5, k = i & 31;
    B4pk[((n >> 4) << 9) + frag_b_idx(n & 15, k)] = (_Float16)wq_val(w4[n * 32 + k], mx[3]);
  }
  for (int i = t; i < 2048; i += 256) {            // fc2: N padded 10->16, K=128
    int n = i >> 7, k = i & 127;
    float v = (n < 10) ? wq_val(fc2w[n * 128 + k], mx[4]) : 0.f;
    Bfc2pk[((k >> 5) << 9) + frag_b_idx(n, k & 31)] = (_Float16)v;
  }
}

// ---------------------------------------------------------------------------
// fc1 weight abs-max (two-pass) + pack
// ---------------------------------------------------------------------------
__global__ void k_absmax_part(const float* __restrict__ w, int n, float* __restrict__ partial) {
  __shared__ float red[256];
  int t = threadIdx.x;
  float m = 0.f;
  for (int i = blockIdx.x * 256 + t; i < n; i += gridDim.x * 256)
    m = fmaxf(m, fabsf(tanhf(w[i])));
  red[t] = m; __syncthreads();
  for (int s = 128; s > 0; s >>= 1) { if (t < s) red[t] = fmaxf(red[t], red[t + s]); __syncthreads(); }
  if (t == 0) partial[blockIdx.x] = red[0];
}

__global__ void k_absmax_final(const float* __restrict__ partial, int n, float* __restrict__ out) {
  __shared__ float red[256];
  int t = threadIdx.x;
  float m = 0.f;
  for (int i = t; i < n; i += 256) m = fmaxf(m, partial[i]);
  red[t] = m; __syncthreads();
  for (int s = 128; s > 0; s >>= 1) { if (t < s) red[t] = fmaxf(red[t], red[t + s]); __syncthreads(); }
  if (t == 0) out[0] = red[0];
}

__global__ void k_pack_bfc1(const float* __restrict__ fc1w, const float* __restrict__ mxp,
                            _Float16* __restrict__ Bpk) {
  float mx = mxp[0];
  const long long total = 128LL * 6272LL;
  for (long long i = blockIdx.x * 256LL + threadIdx.x; i < total; i += (long long)gridDim.x * 256) {
    int n = (int)(i / 6272), k = (int)(i % 6272);
    float v = wq_val(fc1w[i], mx);
    Bpk[(((size_t)(n >> 4) * 196 + (k >> 5)) << 9) + frag_b_idx(n & 15, k & 31)] = (_Float16)v;
  }
}

// ---------------------------------------------------------------------------
// Fused conv1 (C=1 scale) + aq(relu) + shift3x3 + aq(relu) -> packed A for conv2
// ---------------------------------------------------------------------------
__global__ void k_pack_a2(const float* __restrict__ x, const float* __restrict__ wq1,
                          const float* __restrict__ b1, _Float16* __restrict__ Apk) {
  const long long total = 802816LL * 32LL;
  for (long long i = blockIdx.x * 256LL + threadIdx.x; i < total; i += (long long)gridDim.x * 256) {
    int k = (int)(i & 31);
    long long m = i >> 5;           // b*784 + h*28 + w
    float v = 0.f;
    if (k < 16) {
      int b = (int)(m / 784), hw = (int)(m % 784);
      int h = hw / 28, w = hw % 28;
      int g = k % 9;
      int hh = h + g / 3 - 1, ww = w + g % 3 - 1;
      if (hh >= 0 && hh < 28 && ww >= 0 && ww < 28) {
        float y = aq15(x[(size_t)b * 784 + hh * 28 + ww] * wq1[k] + b1[k]);
        v = aq15(y);
      }
    }
    Apk[((size_t)(m >> 4) << 9) + frag_a_idx((int)(m & 15), k)] = (_Float16)v;
  }
}

// shift + aq + maxpool2x2 from conv2 output (f16 row-major [(b,h,w),32])
__global__ void k_pack_a3(const _Float16* __restrict__ Y2, _Float16* __restrict__ Apk) {
  const long long total = 200704LL * 32LL;
  for (long long i = blockIdx.x * 256LL + threadIdx.x; i < total; i += (long long)gridDim.x * 256) {
    int c = (int)(i & 31);
    long long m = i >> 5;           // b*196 + h2*14 + w2
    int b = (int)(m / 196), r = (int)(m % 196);
    int h2 = r / 14, w2 = r % 14;
    int g = c % 9, dh = g / 3 - 1, dw = g % 3 - 1;
    float mv = 0.f;
#pragma unroll
    for (int di = 0; di < 2; ++di)
#pragma unroll
      for (int dj = 0; dj < 2; ++dj) {
        int h = 2 * h2 + di + dh, w = 2 * w2 + dj + dw;
        float s = 0.f;
        if (h >= 0 && h < 28 && w >= 0 && w < 28)
          s = (float)Y2[(((size_t)b * 28 + h) * 28 + w) * 32 + c];
        mv = fmaxf(mv, aq15(s));
      }
    Apk[((size_t)(m >> 4) << 9) + frag_a_idx((int)(m & 15), c)] = (_Float16)mv;
  }
}

// shift + aq from conv3 output (relu already applied)
__global__ void k_pack_a4(const _Float16* __restrict__ Y3, _Float16* __restrict__ Apk) {
  const long long total = 200704LL * 32LL;
  for (long long i = blockIdx.x * 256LL + threadIdx.x; i < total; i += (long long)gridDim.x * 256) {
    int c = (int)(i & 31);
    long long m = i >> 5;           // b*196 + h*14 + w
    int b = (int)(m / 196), r = (int)(m % 196);
    int h = r / 14, w = r % 14;
    int g = c % 9;
    int hh = h + g / 3 - 1, ww = w + g % 3 - 1;
    float s = 0.f;
    if (hh >= 0 && hh < 14 && ww >= 0 && ww < 14)
      s = (float)Y3[(((size_t)b * 14 + hh) * 14 + ww) * 32 + c];
    Apk[((size_t)(m >> 4) << 9) + frag_a_idx((int)(m & 15), c)] = (_Float16)aq15(s);
  }
}

// ---------------------------------------------------------------------------
// Generic WMMA GEMM: one 16x16 tile per wave, register double-buffered K-loop.
// ---------------------------------------------------------------------------
enum { EPI_AQ_F16ROW = 0, EPI_RELU_F16ROW = 1, EPI_PACK_FC1 = 2, EPI_F32ROW = 4 };

template <int MODE>
__global__ void k_gemm(const v16h* __restrict__ Apk, const v16h* __restrict__ Bpk,
                       void* __restrict__ outp, const float* __restrict__ bias,
                       int Mtiles, int Ntiles, int Ktiles, int nBias) {
  int lane = threadIdx.x & 31;
  int tile = blockIdx.x * (blockDim.x >> 5) + (threadIdx.x >> 5);
  if (tile >= Mtiles * Ntiles) return;            // wave-uniform: EXEC stays all-ones
  int mt = tile / Ntiles, nt = tile % Ntiles;
  const v16h* a = Apk + (size_t)mt * Ktiles * 32 + lane;
  const v16h* b = Bpk + (size_t)nt * Ktiles * 32 + lane;
  v8f acc = {};
  // register double-buffer: keep kt+1 loads in flight behind the WMMA on kt
  v16h a0 = a[0];
  v16h b0 = b[0];
  for (int kt = 0; kt < Ktiles; ++kt) {
    v16h a1 = a0, b1 = b0;
    if (kt + 1 < Ktiles) {
      a1 = a[(size_t)(kt + 1) * 32];
      b1 = b[(size_t)(kt + 1) * 32];
      if (kt + 2 < Ktiles) __builtin_prefetch((const void*)(a + (size_t)(kt + 2) * 32), 0, 1);
    }
    acc = __builtin_amdgcn_wmma_f32_16x16x32_f16(false, a0, false, b0, (short)0, acc, false, false);
    a0 = a1; b0 = b1;
  }
  int nl = lane & 15;
  int mh = lane >> 4;
  int n  = nt * 16 + nl;
  float badd = (n < nBias) ? bias[n] : 0.f;
#pragma unroll
  for (int vg = 0; vg < 8; ++vg) {
    int m   = mt * 16 + (vg | (mh << 3));
    float v = acc[vg] + badd;
    if constexpr (MODE == EPI_AQ_F16ROW) {
      ((_Float16*)outp)[(size_t)m * 32 + n] = (_Float16)aq15(v);
    } else if constexpr (MODE == EPI_RELU_F16ROW) {
      ((_Float16*)outp)[(size_t)m * 32 + n] = (_Float16)fmaxf(v, 0.f);
    } else if constexpr (MODE == EPI_PACK_FC1) {
      int bidx = m / 196, hw = m % 196;
      int k = n * 196 + hw;                      // channel-major flatten
      size_t idx = (((size_t)(bidx >> 4) * 196 + (k >> 5)) << 9) + frag_a_idx(bidx & 15, k & 31);
      ((_Float16*)outp)[idx] = (_Float16)aq15(v);
    } else {                                     // EPI_F32ROW: logits, ld = 16
      ((float*)outp)[(size_t)m * 16 + n] = v;
    }
  }
}

// ---------------------------------------------------------------------------
// fc1 GEMM with TDM staging: block = 1 M-tile x all 8 N-tiles (8 waves).
// Wave 0 DMAs the shared A fragment to LDS (read by all 8 waves); every wave
// DMAs its own B fragment. Double-buffered LDS; TENSORcnt + barrier per step.
// M=1024 (64 Mtiles), N=128 (8 Ntiles), K=6272 (196 K-steps).
// ---------------------------------------------------------------------------
__global__ void k_gemm_fc1(const _Float16* __restrict__ Apk, const _Float16* __restrict__ Bpk,
                           _Float16* __restrict__ Ypk, const float* __restrict__ bias) {
  // LDS layout (halfs): A[2][512] at 0, B[2][8][512] at 1024
  __shared__ _Float16 lds[1024 + 8192];
  const int Kt = 196;
  int lane = threadIdx.x & 31;
  int wave = threadIdx.x >> 5;                   // = nt (0..7)
  int mt   = blockIdx.x;                         // 0..63

  const char* Abase = (const char*)(Apk + ((size_t)mt * Kt << 9));
  const char* Bbase = (const char*)(Bpk + ((size_t)wave * Kt << 9));

  // prologue: stage kt=0
  if (wave == 0) stage_1k(Abase, &lds[0], 0u);
  stage_1k(Bbase, &lds[1024 + wave * 512], 2048u + wave * 1024u);
  wait_tdm();
  __syncthreads();

  v8f acc = {};
  for (int kt = 0; kt < Kt; ++kt) {
    int cur = kt & 1, nxt = cur ^ 1;
    if (kt + 1 < Kt) {                           // DMA next buffers (overlaps WMMA below)
      if (wave == 0) stage_1k(Abase + (size_t)(kt + 1) * 1024, &lds[nxt * 512],
                              (unsigned)(nxt * 1024));
      stage_1k(Bbase + (size_t)(kt + 1) * 1024, &lds[1024 + (nxt * 8 + wave) * 512],
               (unsigned)(2048 + (nxt * 8 + wave) * 1024));
    }
    v16h av = *(const v16h*)&lds[cur * 512 + lane * 16];
    v16h bv = *(const v16h*)&lds[1024 + (cur * 8 + wave) * 512 + lane * 16];
    acc = __builtin_amdgcn_wmma_f32_16x16x32_f16(false, av, false, bv, (short)0, acc, false, false);
    wait_tdm();                                  // next buffers resident (+ mem clobber)
    __syncthreads();                             // A(kt+1) visible; reads of cur done
  }

  // epilogue: aq(relu(v + bias)), packed as A-fragments for fc2 (K=128 -> 4 K-tiles)
  int nl = lane & 15;
  int mh = lane >> 4;
  int n  = wave * 16 + nl;
  float badd = bias[n];
#pragma unroll
  for (int vg = 0; vg < 8; ++vg) {
    int m = mt * 16 + (vg | (mh << 3));
    float v = acc[vg] + badd;
    size_t idx = (((size_t)(m >> 4) * 4 + (n >> 5)) << 9) + frag_a_idx(m & 15, n & 31);
    Ypk[idx] = (_Float16)aq15(v);
  }
}

// ---------------------------------------------------------------------------
// log_softmax over 10 classes
// ---------------------------------------------------------------------------
__global__ void k_logsoftmax(const float* __restrict__ logits, float* __restrict__ out) {
  int r = blockIdx.x * blockDim.x + threadIdx.x;
  if (r >= 1024) return;
  const float* z = logits + (size_t)r * 16;
  float mx = -1e30f;
  for (int i = 0; i < 10; ++i) mx = fmaxf(mx, z[i]);
  float s = 0.f;
  for (int i = 0; i < 10; ++i) s += __expf(z[i] - mx);
  float lse = mx + __logf(s);
  for (int i = 0; i < 10; ++i) out[(size_t)r * 10 + i] = z[i] - lse;
}

// ---------------------------------------------------------------------------
// Launch
// ---------------------------------------------------------------------------
extern "C" void kernel_launch(void* const* d_in, const int* in_sizes, int n_in,
                              void* d_out, int out_size, void* d_ws, size_t ws_size,
                              hipStream_t stream) {
  const float* x    = (const float*)d_in[0];
  const float* w1   = (const float*)d_in[1];
  const float* b1   = (const float*)d_in[2];
  const float* w2   = (const float*)d_in[3];
  const float* b2   = (const float*)d_in[4];
  const float* w3   = (const float*)d_in[5];
  const float* b3   = (const float*)d_in[6];
  const float* w4   = (const float*)d_in[7];
  const float* b4   = (const float*)d_in[8];
  const float* fc1w = (const float*)d_in[9];
  const float* fc1b = (const float*)d_in[10];
  const float* fc2w = (const float*)d_in[11];
  const float* fc2b = (const float*)d_in[12];

  const size_t BIG = 51380224;                 // 25,690,112 halfs (one 28x28 stage)
  char* base = (char*)d_ws;
  char* bufA = base;                           // packed-A ping
  char* bufB = base + BIG;                     // GEMM-out / packed pong
  char* sm   = base + 2 * BIG;

  float*    wq1      = (float*)sm;      sm += 256;
  float*    fc1mx    = (float*)sm;      sm += 256;
  float*    partials = (float*)sm;      sm += 4096;
  _Float16* B2pk     = (_Float16*)sm;   sm += 2048;
  _Float16* B3pk     = (_Float16*)sm;   sm += 2048;
  _Float16* B4pk     = (_Float16*)sm;   sm += 2048;
  _Float16* Bfc2pk   = (_Float16*)sm;   sm += 4096;
  _Float16* Bfc1pk   = (_Float16*)sm;   sm += 1605632;   // 8*196 fragments
  _Float16* Yfc1pk   = (_Float16*)sm;   sm += 262144;    // 64*4 fragments
  float*    logits   = (float*)sm;      sm += 65536;     // 1024 x 16 f32

  // Weight preparation
  k_prep_small<<<1, 256, 0, stream>>>(w1, w2, w3, w4, fc2w, wq1, B2pk, B3pk, B4pk, Bfc2pk);
  k_absmax_part<<<784, 256, 0, stream>>>(fc1w, 802816, partials);
  k_absmax_final<<<1, 256, 0, stream>>>(partials, 784, fc1mx);
  k_pack_bfc1<<<1024, 256, 0, stream>>>(fc1w, fc1mx, Bfc1pk);

  // conv1 + aq + shift + aq, packed for conv2
  k_pack_a2<<<8192, 256, 0, stream>>>(x, wq1, b1, (_Float16*)bufA);

  // conv2: [802816,32pad] x [32,32] -> aq(relu) f16 row-major
  k_gemm<EPI_AQ_F16ROW><<<(50176 * 2) / 8, 256, 0, stream>>>(
      (const v16h*)bufA, (const v16h*)B2pk, bufB, b2, 50176, 2, 1, 32);

  // shift + aq + maxpool -> packed A for conv3
  k_pack_a3<<<4096, 256, 0, stream>>>((const _Float16*)bufB, (_Float16*)bufA);

  // conv3: relu epilogue
  k_gemm<EPI_RELU_F16ROW><<<(12544 * 2) / 8, 256, 0, stream>>>(
      (const v16h*)bufA, (const v16h*)B3pk, bufB, b3, 12544, 2, 1, 32);

  // shift + aq -> packed A for conv4
  k_pack_a4<<<4096, 256, 0, stream>>>((const _Float16*)bufB, (_Float16*)bufA);

  // conv4: aq(relu) + channel-major flatten packed into fc1 A fragments
  k_gemm<EPI_PACK_FC1><<<(12544 * 2) / 8, 256, 0, stream>>>(
      (const v16h*)bufA, (const v16h*)B4pk, bufB, b4, 12544, 2, 1, 32);

  // fc1 via TDM-staged LDS: 64 blocks x (1 Mtile x 8 Ntiles)
  k_gemm_fc1<<<64, 256, 0, stream>>>(
      (const _Float16*)bufB, Bfc1pk, Yfc1pk, fc1b);

  // fc2: M=1024, N=16(pad of 10), K=128 -> f32 logits
  k_gemm<EPI_F32ROW><<<(64 * 1) / 8, 256, 0, stream>>>(
      (const v16h*)Yfc1pk, (const v16h*)Bfc2pk, logits, fc2b, 64, 1, 4, 10);

  // log_softmax -> d_out [1024,10] f32
  k_logsoftmax<<<4, 256, 0, stream>>>(logits, (float*)d_out);
}
